// TensorBase_65738769432614
// MI455X (gfx1250) — matrix operand order repaired and checked by reading, not compiled, not run on hardware
//
#include <hip/hip_runtime.h>

// ---------------- problem constants (from reference) ----------------
#define RESG   300          // grid resolution
#define NCH    64           // NC_A + NC_D channels
#define NCA    48           // app channels
#define KDIM   144          // 3 * NCA
#define KPAD   160          // K padded to 5 blocks of 32 for wmma f16
#define APPD   27           // app_dim
#define WB_STRIDE 170       // LDS row stride (halves): avoids bank conflicts

typedef __attribute__((ext_vector_type(16))) _Float16 v16h;
typedef __attribute__((ext_vector_type(8)))  float    v8f;

union V16H { v16h v; unsigned u[8]; };
union V8F  { v8f  v; float    f[8]; };
union HPK  { _Float16 h[2]; unsigned u; };

// ------------------------------------------------------------------
// Pre-pass: transpose plane_coef (3,64,300,300) -> (3,300,300,64)
// so the main kernel's 4-corner gathers become coalesced 128B reads.
// ------------------------------------------------------------------
__global__ __launch_bounds__(256) void transpose_planes(
    const float* __restrict__ src, float* __restrict__ dst)
{
    const size_t total = 3ull * RESG * RESG * NCH;   // 17,280,000
    for (size_t i = (size_t)blockIdx.x * 256 + threadIdx.x; i < total;
         i += (size_t)gridDim.x * 256) {
        int    c  = (int)(i & (NCH - 1));
        size_t t  = i >> 6;                // p*RES*RES + y*RES + x
        int    x  = (int)(t % RESG);
        size_t t2 = t / RESG;
        int    y  = (int)(t2 % RESG);
        int    p  = (int)(t2 / RESG);
        dst[i] = src[(((size_t)p * NCH + c) * RESG + y) * RESG + x];
    }
}

// ------------------------------------------------------------------
// Fused kernel: gather + line interp + sigma + WMMA basis projection.
// Block = 256 threads = 8 waves; each wave owns 16 samples.
// Lane -> channel pair (2*lane, 2*lane+1): one float2 load per corner.
// ------------------------------------------------------------------
__global__ __launch_bounds__(256) void tensorf_fused(
    const float* __restrict__ xyz,     // (N,3)
    const float* __restrict__ planes,  // transposed (p,y,x,c) or orig (p,c,y,x)
    const float* __restrict__ lines,   // (3,64,300,1)
    const float* __restrict__ bw,      // (27,144)
    float* __restrict__ outSigma,      // (N,)
    float* __restrict__ outApp,        // (N,27)
    int nSamples, int transposed)
{
    __shared__ _Float16 sWbT[32][WB_STRIDE];        // basis, [n][k] transposed
    __shared__ _Float16 sFeat[8][16][WB_STRIDE];    // per-wave 16 x KPAD f16 tile

    const int tid  = threadIdx.x;
    const int wave = tid >> 5;
    const int lane = tid & 31;
    const int half = lane >> 4;
    const int l16  = lane & 15;

    // ---- build transposed basis in LDS (zero-padded to 32 x KPAD) ----
    for (int idx = tid; idx < 32 * KPAD; idx += 256) {
        int n = idx / KPAD, k = idx % KPAD;
        float v = (n < APPD && k < KDIM) ? bw[n * KDIM + k] : 0.0f;
        sWbT[n][k] = (_Float16)v;
    }
    // ---- zero K-padding columns of this wave's feature tile ----
    for (int i = lane; i < 16 * (KPAD - KDIM); i += 32) {
        int m = i >> 4, k = KDIM + (i & 15);
        sFeat[wave][m][k] = (_Float16)0.0f;
    }
    __syncthreads();

    const int base = (blockIdx.x * 8 + wave) * 16;
    const int c1 = 2 * lane;        // even channel 0..62
    const int c2 = 2 * lane + 1;    // odd channel 1..63  (48..63 -> lanes 24..31)

    const int mat0[3] = {0, 0, 1};
    const int mat1[3] = {1, 2, 2};
    const int vecm[3] = {2, 1, 0};

    // =================== gather phase ===================
    for (int sm = 0; sm < 16; ++sm) {
        int samp = base + sm;
        int s    = samp < nSamples ? samp : nSamples - 1;   // clamp for loads
        float co[3];
        co[0] = xyz[(size_t)s * 3 + 0];
        co[1] = xyz[(size_t)s * 3 + 1];
        co[2] = xyz[(size_t)s * 3 + 2];

        float dpart = 0.0f;   // density partial (lanes 24..31 contribute)

        #pragma unroll
        for (int p = 0; p < 3; ++p) {
            float gx = co[mat0[p]], gy = co[mat1[p]], gz = co[vecm[p]];
            float ix = (gx + 1.0f) * 0.5f * (RESG - 1);
            float iy = (gy + 1.0f) * 0.5f * (RESG - 1);
            float iz = (gz + 1.0f) * 0.5f * (RESG - 1);
            float x0f = floorf(ix), y0f = floorf(iy), z0f = floorf(iz);
            float wx = ix - x0f, wy = iy - y0f, wz = iz - z0f;
            int x0 = (int)x0f; x0 = x0 < 0 ? 0 : (x0 > RESG - 1 ? RESG - 1 : x0);
            int y0 = (int)y0f; y0 = y0 < 0 ? 0 : (y0 > RESG - 1 ? RESG - 1 : y0);
            int z0 = (int)z0f; z0 = z0 < 0 ? 0 : (z0 > RESG - 1 ? RESG - 1 : z0);
            int x1 = x0 + 1 < RESG ? x0 + 1 : RESG - 1;   // wx==0 when clamped
            int y1 = y0 + 1 < RESG ? y0 + 1 : RESG - 1;
            int z1 = z0 + 1 < RESG ? z0 + 1 : RESG - 1;
            float w00 = (1.0f - wx) * (1.0f - wy);
            float w01 = wx * (1.0f - wy);
            float w10 = (1.0f - wx) * wy;
            float w11 = wx * wy;

            float v00a, v01a, v10a, v11a, v00b, v01b, v10b, v11b;
            if (transposed) {
                // channel-last: one aligned float2 per corner per lane,
                // wave covers all 64 channels of a corner in one b64 load.
                const float* pb = planes + (size_t)p * RESG * RESG * NCH;
                size_t o00 = ((size_t)y0 * RESG + x0) * NCH + c1;
                size_t o01 = ((size_t)y0 * RESG + x1) * NCH + c1;
                size_t o10 = ((size_t)y1 * RESG + x0) * NCH + c1;
                size_t o11 = ((size_t)y1 * RESG + x1) * NCH + c1;
                float2 q00 = *(const float2*)(pb + o00);
                float2 q01 = *(const float2*)(pb + o01);
                float2 q10 = *(const float2*)(pb + o10);
                float2 q11 = *(const float2*)(pb + o11);
                v00a = q00.x; v00b = q00.y;
                v01a = q01.x; v01b = q01.y;
                v10a = q10.x; v10b = q10.y;
                v11a = q11.x; v11b = q11.y;
            } else {
                // original (p,c,y,x) layout fallback (uncoalesced, L2-resident)
                const float* pb = planes + (size_t)p * NCH * RESG * RESG;
                size_t cha = (size_t)c1 * RESG * RESG;
                size_t chb = (size_t)c2 * RESG * RESG;
                size_t r0  = (size_t)y0 * RESG, r1 = (size_t)y1 * RESG;
                v00a = pb[cha + r0 + x0]; v00b = pb[chb + r0 + x0];
                v01a = pb[cha + r0 + x1]; v01b = pb[chb + r0 + x1];
                v10a = pb[cha + r1 + x0]; v10b = pb[chb + r1 + x0];
                v11a = pb[cha + r1 + x1]; v11b = pb[chb + r1 + x1];
            }
            float pf1 = w00 * v00a + w01 * v01a + w10 * v10a + w11 * v11a;
            float pf2 = w00 * v00b + w01 * v01b + w10 * v10b + w11 * v11b;

            // line interp along H (W==1, x coord fixed at 0)
            const float* lb = lines + (size_t)p * NCH * RESG;
            float lf1 = (1.0f - wz) * lb[(size_t)c1 * RESG + z0] + wz * lb[(size_t)c1 * RESG + z1];
            float lf2 = (1.0f - wz) * lb[(size_t)c2 * RESG + z0] + wz * lb[(size_t)c2 * RESG + z1];

            float f1 = pf1 * lf1;
            float f2 = pf2 * lf2;

            if (c1 < NCA) {
                // app channels: packed f16 pair, single aligned b32 LDS store
                HPK pk;
                pk.h[0] = (_Float16)f1;
                pk.h[1] = (_Float16)f2;
                *(unsigned*)&sFeat[wave][sm][p * NCA + c1] = pk.u;
            } else {
                dpart += f1 + f2;   // density channels 48..63 (lanes 24..31)
            }
        }
        // wave32 reduction of density partials
        #pragma unroll
        for (int m = 16; m >= 1; m >>= 1) dpart += __shfl_xor(dpart, m, 32);
        if (lane == 0 && samp < nSamples) outSigma[samp] = dpart;
    }
    __syncthreads();

    // =================== WMMA projection phase ===================
    // D(16x16) tiles: cols 0..15 and 16..26 of the 27-wide output.
    V8F acc0, acc1;
    #pragma unroll
    for (int r = 0; r < 8; ++r) { acc0.f[r] = 0.0f; acc1.f[r] = 0.0f; }

    const _Float16* frow  = sFeat[wave][l16];   // A: row m = lane&15
    const _Float16* brow0 = sWbT[l16];          // B tile 0: col n = lane&15
    const _Float16* brow1 = sWbT[l16 + 16];     // B tile 1: col n+16

    #pragma unroll
    for (int kb = 0; kb < KPAD / 32; ++kb) {
        V16H A, B0, B1;
        const int ka  = kb * 32 + half * 8;    // A lane K-base (ISA 16-bit A layout)
        const int kbb = kb * 32 + half * 16;   // unused: B reads use ka, same k order as A
        #pragma unroll
        for (int v = 0; v < 8; ++v) {
            int kA = ka + ((v < 4) ? v * 2 : 16 + (v - 4) * 2);
            int kB = ka + ((v < 4) ? v * 2 : 16 + (v - 4) * 2);
            A.u[v]  = *(const unsigned*)(frow  + kA);   // {K, K+1} packed halves
            B0.u[v] = *(const unsigned*)(brow0 + kB);
            B1.u[v] = *(const unsigned*)(brow1 + kB);
        }
        acc0.v = __builtin_amdgcn_wmma_f32_16x16x32_f16(
            false, A.v, false, B0.v, (short)0, acc0.v, false, false);
        acc1.v = __builtin_amdgcn_wmma_f32_16x16x32_f16(
            false, A.v, false, B1.v, (short)0, acc1.v, false, false);
    }

    // D layout: VGPR r -> M = r + 8*half, N = lane&15
    #pragma unroll
    for (int r = 0; r < 8; ++r) {
        int m    = r + 8 * half;
        int samp = base + m;
        if (samp < nSamples) {
            outApp[(size_t)samp * APPD + l16] = acc0.f[r];
            if (l16 < APPD - 16)
                outApp[(size_t)samp * APPD + 16 + l16] = acc1.f[r];
        }
    }
}

// ------------------------------------------------------------------
extern "C" void kernel_launch(void* const* d_in, const int* in_sizes, int n_in,
                              void* d_out, int out_size, void* d_ws, size_t ws_size,
                              hipStream_t stream) {
    (void)n_in; (void)out_size;
    const float* xyz   = (const float*)d_in[0];
    const float* plane = (const float*)d_in[1];
    const float* line  = (const float*)d_in[2];
    const float* bw    = (const float*)d_in[3];

    const int nSamples = in_sizes[0] / 3;
    float* outSigma = (float*)d_out;
    float* outApp   = (float*)d_out + nSamples;

    const size_t need = 3ull * RESG * RESG * NCH * sizeof(float);   // ~69 MB
    const int transposed = (ws_size >= need) ? 1 : 0;

    if (transposed) {
        const size_t total = 3ull * RESG * RESG * NCH;
        int tgrid = (int)((total + 255) / 256);
        transpose_planes<<<tgrid, 256, 0, stream>>>(plane, (float*)d_ws);
    }
    const float* psrc = transposed ? (const float*)d_ws : plane;

    int grid = (nSamples + 127) / 128;   // 128 samples per block (8 waves x 16)
    tensorf_fused<<<grid, 256, 0, stream>>>(xyz, psrc, line, bw,
                                            outSigma, outApp, nSamples, transposed);
}
